// MultiHeadAttention_2920577761759
// MI455X (gfx1250) — matrix operand order repaired and checked by reading, not compiled, not run on hardware
//
#include <hip/hip_runtime.h>

typedef __bf16        v16bf __attribute__((ext_vector_type(16)));
typedef float         v8f   __attribute__((ext_vector_type(8)));
typedef unsigned int  v4u   __attribute__((ext_vector_type(4)));

#define D_MODEL 1024
#define S_LEN   2048
#define BATCH   4
#define HEADS   16
#define DEPTH   64

// Native bf16 converts (RNE) -> lets backend use v_cvt_pk_bf16_f32 class ops.
__device__ __forceinline__ unsigned short f2bf(float f) {
  union { __bf16 b; unsigned short u; } t;
  t.b = (__bf16)f;
  return t.u;
}

__device__ __forceinline__ unsigned int pack2(float lo, float hi) {
  union { __bf16 b[2]; unsigned int u; } t;
  t.b[0] = (__bf16)lo;
  t.b[1] = (__bf16)hi;
  return t.u;
}

__device__ __forceinline__ v4u pack8(float4 a, float4 b) {
  v4u r;
  r[0] = pack2(a.x, a.y); r[1] = pack2(a.z, a.w);
  r[2] = pack2(b.x, b.y); r[3] = pack2(b.z, b.w);
  return r;
}

__device__ __forceinline__ v8f wmma_bf16(v16bf a, v16bf b, v8f c) {
  // D = A(16x32 bf16) x B(32x16 bf16) + C(16x16 f32)
  return __builtin_amdgcn_wmma_f32_16x16x32_bf16(
      /*neg_a=*/false, a, /*neg_b=*/false, b,
      /*c_mod=*/(short)0, c, /*reuse_a=*/false, /*reuse_b=*/false);
}

// A fragment 16x32 (MxK) from LDS [row][k], stride ld (ushorts, 16B-divisible).
// Lanes 0-15: M=0..15, K {0..7,16..23}; lanes 16-31: M=0..15, K {8..15,24..31}.
// Each half is a contiguous 16B run -> 2x ds_load_b128.
__device__ __forceinline__ v16bf load_a_frag(const unsigned short* lds,
                                             int rowBase, int kBase, int ld) {
  const int lane = threadIdx.x & 31;
  const int r = rowBase + (lane & 15);
  const int klo = (lane < 16) ? 0 : 8;
  const v4u* p = reinterpret_cast<const v4u*>(lds + r * ld + kBase + klo);
  union { v16bf v; v4u q[2]; } f;
  f.q[0] = p[0];   // K klo+0..7
  f.q[1] = p[2];   // K klo+16..23 (+16 ushorts = +2 v4u)
  return f.v;
}

// B fragment 32x16 (KxN) from LDS [n][k] layout (stride ld, 16B-divisible).
// Lanes 0-15: N, K {0..7,16..23}; lanes 16-31: N, K {8..15,24..31}.
__device__ __forceinline__ v16bf load_b_frag(const unsigned short* lds,
                                             int nBase, int kBase, int ld) {
  const int lane = threadIdx.x & 31;
  const int n = nBase + (lane & 15);
  const v4u* p = reinterpret_cast<const v4u*>(lds + n * ld + kBase +
                                              ((lane < 16) ? 0 : 8));
  union { v16bf v; v4u q[2]; } f;
  f.q[0] = p[0];
  f.q[1] = p[2];
  return f.v;
}

// ---------------------------------------------------------------------------
// GEMM: C[M,N] = A[M,K] @ W[N,K]^T + bias[N]   (M=8192, N=K=1024)
// OUT_MODE 0: fp32 row-major [M,N]
// OUT_MODE 1: bf16 split-head [B,H,S,DEPTH]
// OUT_MODE 2: bf16 split-head transposed [B,H,DEPTH,S]   (for V)
// ---------------------------------------------------------------------------
template <int OUT_MODE>
__global__ __launch_bounds__(256) void gemm_bias_kernel(
    const float* __restrict__ A, const float* __restrict__ W,
    const float* __restrict__ bias, void* __restrict__ out) {
  constexpr int BM = 128, BN = 64, BK = 64, LD = 72;
  __shared__ __align__(16) unsigned short sA[BM * LD];
  __shared__ __align__(16) unsigned short sB[BN * LD];

  const int t = threadIdx.x;
  const int lane = t & 31, wave = t >> 5;
  const int wm = wave >> 1, wn = wave & 1;  // 4 x 2 wave grid
  const int blockN = blockIdx.x * BN;
  const int blockM = blockIdx.y * BM;

  v8f acc[2][2];
#pragma unroll
  for (int i = 0; i < 2; ++i)
#pragma unroll
    for (int j = 0; j < 2; ++j) acc[i][j] = {};

  const int arow = t >> 1, acb = (t & 1) * 32;  // A: 128 rows x 64 cols
  const int wrow = t >> 2, wcb = (t & 3) * 16;  // W: 64 rows x 64 cols

  for (int k0 = 0; k0 < D_MODEL; k0 += BK) {
    {  // stage A slab fp32->bf16, packed b128 stores
      const float4* src = reinterpret_cast<const float4*>(
          A + (size_t)(blockM + arow) * D_MODEL + k0 + acb);
      v4u* dst = reinterpret_cast<v4u*>(&sA[arow * LD + acb]);
#pragma unroll
      for (int j = 0; j < 4; ++j) dst[j] = pack8(src[2 * j], src[2 * j + 1]);
    }
    {  // stage W slab
      const float4* src = reinterpret_cast<const float4*>(
          W + (size_t)(blockN + wrow) * D_MODEL + k0 + wcb);
      v4u* dst = reinterpret_cast<v4u*>(&sB[wrow * LD + wcb]);
#pragma unroll
      for (int j = 0; j < 2; ++j) dst[j] = pack8(src[2 * j], src[2 * j + 1]);
    }
    if (k0 + BK < D_MODEL)  // hint next slab (global_prefetch_b8)
      __builtin_prefetch(A + (size_t)(blockM + arow) * D_MODEL + k0 + BK + acb, 0, 1);
    __syncthreads();

#pragma unroll
    for (int kk = 0; kk < BK; kk += 32) {
      v16bf a0 = load_a_frag(sA, wm * 32, kk, LD);
      v16bf a1 = load_a_frag(sA, wm * 32 + 16, kk, LD);
      v16bf b0 = load_b_frag(sB, wn * 32, kk, LD);
      v16bf b1 = load_b_frag(sB, wn * 32 + 16, kk, LD);
      acc[0][0] = wmma_bf16(a0, b0, acc[0][0]);
      acc[0][1] = wmma_bf16(a0, b1, acc[0][1]);
      acc[1][0] = wmma_bf16(a1, b0, acc[1][0]);
      acc[1][1] = wmma_bf16(a1, b1, acc[1][1]);
    }
    __syncthreads();
  }

  // epilogue: C layout lane l, VGPR r -> M = r + 8*(l/16), N = l%16
#pragma unroll
  for (int mt = 0; mt < 2; ++mt)
#pragma unroll
    for (int nt = 0; nt < 2; ++nt)
#pragma unroll
      for (int r = 0; r < 8; ++r) {
        int gm = blockM + wm * 32 + mt * 16 + r + 8 * (lane >> 4);
        int gn = blockN + wn * 32 + nt * 16 + (lane & 15);
        float v = acc[mt][nt][r] + bias[gn];
        int bi = gm >> 11, s = gm & (S_LEN - 1);
        int hh = gn >> 6, d = gn & (DEPTH - 1);
        if (OUT_MODE == 1) {
          ((unsigned short*)out)[((size_t)(bi * HEADS + hh) * S_LEN + s) * DEPTH + d] =
              f2bf(v);
        } else if (OUT_MODE == 2) {
          ((unsigned short*)out)[((size_t)(bi * HEADS + hh) * DEPTH + d) * S_LEN + s] =
              f2bf(v);
        } else {
          ((float*)out)[(size_t)gm * D_MODEL + gn] = v;
        }
      }
}

// ---------------------------------------------------------------------------
// Fused attention (flash-style): grid (S/64, B*H), 128 threads = 4 waves.
// Q,K bf16 [B,H,S,64]; V bf16 transposed [B,H,64,S]. Online softmax per row,
// carried in the base-2 domain (temperature folds in log2(e)).
// ---------------------------------------------------------------------------
__global__ __launch_bounds__(128) void flash_attn_kernel(
    const unsigned short* __restrict__ Q, const unsigned short* __restrict__ K,
    const unsigned short* __restrict__ V, float* __restrict__ O) {
  constexpr int LD = 72;
  __shared__ __align__(16) unsigned short sQ[64 * LD];
  __shared__ __align__(16) unsigned short sK[64 * LD];  // [key][depth]
  __shared__ __align__(16) unsigned short sV[64 * LD];  // [depth][key]
  __shared__ __align__(16) unsigned short sP[64 * LD];  // [q][key]

  const int t = threadIdx.x;
  const int lane = t & 31, wave = t >> 5;
  const int bh = blockIdx.y;
  const int b = bh >> 4, h = bh & 15;
  const int qBase = blockIdx.x * 64;
  const size_t headOff = (size_t)bh * S_LEN * DEPTH;

  const int row = t >> 1, cb = (t & 1) * 32;  // 64 rows x 64 cols staging
  {  // stage Q chunk once (b128 copies)
    const v4u* src = reinterpret_cast<const v4u*>(
        Q + headOff + (size_t)(qBase + row) * DEPTH + cb);
    v4u* dst = reinterpret_cast<v4u*>(&sQ[row * LD + cb]);
#pragma unroll
    for (int j = 0; j < 4; ++j) dst[j] = src[j];
  }
  __syncthreads();

  const v16bf aq0 = load_a_frag(sQ, wave * 16, 0, LD);
  const v16bf aq1 = load_a_frag(sQ, wave * 16, 32, LD);

  v8f o[4];
  float m_r[8], l_r[8];
#pragma unroll
  for (int a = 0; a < 4; ++a) o[a] = {};
#pragma unroll
  for (int r = 0; r < 8; ++r) { m_r[r] = -1e30f; l_r[r] = 0.f; }

  // exp2((x - m) * sm * log2e) == exp((x - m) * sm); fold into one constant
  const float sml2 = 0.125f * 1.4426950408889634f;

  for (int kt = 0; kt < S_LEN / 64; ++kt) {
    const int keyBase = kt * 64;
    {  // stage K [key][depth] and V^T [depth][key] tiles, all b128
      const v4u* ks = reinterpret_cast<const v4u*>(
          K + headOff + (size_t)(keyBase + row) * DEPTH + cb);
      v4u* kd = reinterpret_cast<v4u*>(&sK[row * LD + cb]);
      const v4u* vs = reinterpret_cast<const v4u*>(
          V + headOff + (size_t)row * S_LEN + keyBase + cb);
      v4u* vd = reinterpret_cast<v4u*>(&sV[row * LD + cb]);
#pragma unroll
      for (int j = 0; j < 4; ++j) { kd[j] = ks[j]; vd[j] = vs[j]; }
    }
    __syncthreads();

    // S = Q @ K^T  (16q x 64k per wave; contraction over depth=64)
    v8f sacc[4];
#pragma unroll
    for (int a = 0; a < 4; ++a) sacc[a] = {};
#pragma unroll
    for (int nt = 0; nt < 4; ++nt) {
      v16bf bk0 = load_b_frag(sK, nt * 16, 0, LD);
      v16bf bk1 = load_b_frag(sK, nt * 16, 32, LD);
      sacc[nt] = wmma_bf16(aq0, bk0, sacc[nt]);
      sacc[nt] = wmma_bf16(aq1, bk1, sacc[nt]);
    }

    // online softmax (base-2); a C-layout row lives in one 16-lane half
    const int prow = wave * 16 + 8 * (lane >> 4);
    const int pcol = lane & 15;
#pragma unroll
    for (int r = 0; r < 8; ++r) {
      float tmax = -1e30f;
#pragma unroll
      for (int a = 0; a < 4; ++a) tmax = fmaxf(tmax, sacc[a][r]);
      tmax *= sml2;
#pragma unroll
      for (int msk = 8; msk >= 1; msk >>= 1)
        tmax = fmaxf(tmax, __shfl_xor(tmax, msk, 32));
      float mnew = fmaxf(m_r[r], tmax);
      float corr = exp2f(m_r[r] - mnew);
      float lsum = 0.f;
#pragma unroll
      for (int a = 0; a < 4; ++a) {
        float p = exp2f(sacc[a][r] * sml2 - mnew);
        lsum += p;
        sP[(prow + r) * LD + a * 16 + pcol] = f2bf(p);
        o[a][r] *= corr;
      }
#pragma unroll
      for (int msk = 8; msk >= 1; msk >>= 1) lsum += __shfl_xor(lsum, msk, 32);
      l_r[r] = l_r[r] * corr + lsum;
      m_r[r] = mnew;
    }

    // O += P @ V  (contraction over 64 keys; sV is [d][key] -> nk layout)
    v16bf ap0 = load_a_frag(sP, wave * 16, 0, LD);
    v16bf ap1 = load_a_frag(sP, wave * 16, 32, LD);
#pragma unroll
    for (int nt = 0; nt < 4; ++nt) {
      v16bf bv0 = load_b_frag(sV, nt * 16, 0, LD);
      v16bf bv1 = load_b_frag(sV, nt * 16, 32, LD);
      o[nt] = wmma_bf16(ap0, bv0, o[nt]);
      o[nt] = wmma_bf16(ap1, bv1, o[nt]);
    }
    __syncthreads();
  }

  // normalize and write fp32 [B,S,D_MODEL] (merge heads)
#pragma unroll
  for (int nt = 0; nt < 4; ++nt)
#pragma unroll
    for (int r = 0; r < 8; ++r) {
      int rr = wave * 16 + r + 8 * (lane >> 4);
      int d = nt * 16 + (lane & 15);
      O[((size_t)b * S_LEN + (qBase + rr)) * D_MODEL + h * DEPTH + d] =
          o[nt][r] * (1.0f / l_r[r]);
    }
}

extern "C" void kernel_launch(void* const* d_in, const int* in_sizes, int n_in,
                              void* d_out, int out_size, void* d_ws, size_t ws_size,
                              hipStream_t stream) {
  const float* v_in = (const float*)d_in[0];
  const float* k_in = (const float*)d_in[1];
  const float* q_in = (const float*)d_in[2];
  const float* wq_w = (const float*)d_in[3];
  const float* wq_b = (const float*)d_in[4];
  const float* wk_w = (const float*)d_in[5];
  const float* wk_b = (const float*)d_in[6];
  const float* wv_w = (const float*)d_in[7];
  const float* wv_b = (const float*)d_in[8];
  const float* dw   = (const float*)d_in[9];
  const float* db   = (const float*)d_in[10];

  // workspace: Q/K bf16 [B,H,S,D], V bf16 [B,H,D,S] (16.8 MB each) + fp32 attn
  const size_t headElems = (size_t)BATCH * HEADS * S_LEN * DEPTH;
  unsigned short* qb = (unsigned short*)d_ws;
  unsigned short* kb = qb + headElems;
  unsigned short* vb = kb + headElems;
  float* attn = (float*)(vb + headElems);

  dim3 gg(D_MODEL / 64, (BATCH * S_LEN) / 128);  // (16, 64)
  gemm_bias_kernel<1><<<gg, 256, 0, stream>>>(q_in, wq_w, wq_b, (void*)qb);
  gemm_bias_kernel<1><<<gg, 256, 0, stream>>>(k_in, wk_w, wk_b, (void*)kb);
  gemm_bias_kernel<2><<<gg, 256, 0, stream>>>(v_in, wv_w, wv_b, (void*)vb);

  dim3 ga(S_LEN / 64, BATCH * HEADS);            // (32, 64)
  flash_attn_kernel<<<ga, 128, 0, stream>>>(qb, kb, vb, attn);

  gemm_bias_kernel<0><<<gg, 256, 0, stream>>>(attn, dw, db, d_out);
}
